// Seq2Seq_36971078484642
// MI455X (gfx1250) — matrix operand order repaired and checked by reading, not compiled, not run on hardware
//
#include <hip/hip_runtime.h>
#include <hip/hip_bf16.h>

// ---------------------------------------------------------------------------
// Problem constants (reference shapes are static)
// ---------------------------------------------------------------------------
#define BB   32
#define TT   48
#define LL   400
#define H2   512
#define EE   128
#define VV   50000
#define VEXT 50100
#define BT   (BB * TT)      // 1536
#define CATW (3 * H2)       // 1536

typedef __attribute__((ext_vector_type(16))) __bf16 v16bf;
typedef __attribute__((ext_vector_type(8)))  float  v8f;

// ---------------------------------------------------------------------------
// Fragment packing: zero-padded bf16 blocks of 512 elements per (tile, kstep).
// Lane l owns bf16[ l*16 .. l*16+15 ] of each block -> wave reads a coalesced,
// 32B-aligned 1KB block; GEMM inner loop has NO bounds checks / converts.
//
// A-fragment element map (16x32, contiguous k order, same as B): row = mTile*16 + (lane&15),
//   k = s*32 + (lane>>4)*16 + i
// B-fragment element map (32x16): col = nTile*16 + (lane&15),
//   k = s*32 + (lane>>4)*16 + i
// ---------------------------------------------------------------------------
__global__ void pack_a_kernel(const float* __restrict__ A, __bf16* __restrict__ P,
                              int M, int K, int lda, long long strideA,
                              int mT, int S, long long total) {
  long long idx = (long long)blockIdx.x * blockDim.x + threadIdx.x;
  if (idx >= total) return;
  int i    = (int)(idx & 15);
  int lane = (int)((idx >> 4) & 31);
  long long bms = idx >> 9;                  // ((batch*mT)+mt)*S + s
  int s     = (int)(bms % S);
  int mt    = (int)((bms / S) % mT);
  int batch = (int)(bms / ((long long)S * mT));
  int row = mt * 16 + (lane & 15);
  int k   = s * 32 + ((lane >> 4) << 4) + i;
  float v = (row < M && k < K) ? A[(size_t)batch * strideA + (size_t)row * lda + k] : 0.0f;
  P[idx] = (__bf16)v;
}

// B source: fp32 row-major (K x N), leading dim ldb
__global__ void pack_b_kernel(const float* __restrict__ B, __bf16* __restrict__ P,
                              int K, int N, int ldb, long long strideB,
                              int nT, int S, long long total) {
  long long idx = (long long)blockIdx.x * blockDim.x + threadIdx.x;
  if (idx >= total) return;
  int i    = (int)(idx & 15);
  int lane = (int)((idx >> 4) & 31);
  long long bns = idx >> 9;
  int s     = (int)(bns % S);
  int nt    = (int)((bns / S) % nT);
  int batch = (int)(bns / ((long long)S * nT));
  int col = nt * 16 + (lane & 15);
  int k   = s * 32 + ((lane >> 4) << 4) + i;
  float v = (col < N && k < K) ? B[(size_t)batch * strideB + (size_t)k * ldb + col] : 0.0f;
  P[idx] = (__bf16)v;
}

// B source: fp32 "transposed" (N x K) row-major -> contiguous per-lane gather
__global__ void pack_bT_kernel(const float* __restrict__ Bt, __bf16* __restrict__ P,
                               int K, int N, int ldb, long long strideB,
                               int nT, int S, long long total) {
  long long idx = (long long)blockIdx.x * blockDim.x + threadIdx.x;
  if (idx >= total) return;
  int i    = (int)(idx & 15);
  int lane = (int)((idx >> 4) & 31);
  long long bns = idx >> 9;
  int s     = (int)(bns % S);
  int nt    = (int)((bns / S) % nT);
  int batch = (int)(bns / ((long long)S * nT));
  int col = nt * 16 + (lane & 15);
  int k   = s * 32 + ((lane >> 4) << 4) + i;
  float v = (col < N && k < K) ? Bt[(size_t)batch * strideB + (size_t)col * ldb + k] : 0.0f;
  P[idx] = (__bf16)v;
}

// ---------------------------------------------------------------------------
// Branch-free batched WMMA GEMM over packed operands: wave per 16x16 C tile.
// Inner loop: 2x 32B loads + v_wmma_f32_16x16x32_bf16, f32 accumulate.
// M, N are exact tile multiples in every call -> unconditional stores.
// ---------------------------------------------------------------------------
__global__ void gemm_wmma_kernel(const __bf16* __restrict__ Ap,
                                 const __bf16* __restrict__ Bp,
                                 float* __restrict__ C,
                                 int mT, int nT, int S, int ldc,
                                 long long strideC, int batches) {
  int gw   = (int)((blockIdx.x * blockDim.x + threadIdx.x) >> 5);
  int lane = threadIdx.x & 31;
  int tiles = mT * nT;
  if (gw >= batches * tiles) return;           // wave-uniform exit (EXEC stays full)
  int batch = gw / tiles;
  int rem   = gw % tiles;
  int mTile = rem / nT;
  int nTile = rem % nT;

  const __bf16* a = Ap + ((size_t)((size_t)batch * mT + mTile) * S) * 512 + (size_t)lane * 16;
  const __bf16* b = Bp + ((size_t)((size_t)batch * nT + nTile) * S) * 512 + (size_t)lane * 16;

  v8f acc = {};
  for (int s = 0; s < S; ++s) {
    v16bf av = *(const v16bf*)a;
    v16bf bv = *(const v16bf*)b;
    a += 512; b += 512;
    acc = __builtin_amdgcn_wmma_f32_16x16x32_bf16(false, av, false, bv,
                                                  (short)0, acc, false, false);
  }

  float* Cb = C + (size_t)batch * strideC;
  int colC  = nTile * 16 + (lane & 15);
  int rbase = mTile * 16 + ((lane >> 4) << 3);
#pragma unroll
  for (int j = 0; j < 8; ++j)
    Cb[(size_t)(rbase + j) * ldc + colC] = acc[j];
}

// ---------------------------------------------------------------------------
// Intra-temporal attention: one thread per (b,l) column, sequential cumsum in t.
// ---------------------------------------------------------------------------
__global__ void temporal_kernel(const float* __restrict__ e, float* __restrict__ att) {
  int idx = blockIdx.x * blockDim.x + threadIdx.x;
  if (idx >= BB * LL) return;
  int b = idx / LL, l = idx % LL;
  float acc = 0.0f;
  for (int t = 0; t < TT; ++t) {
    size_t o = ((size_t)b * TT + t) * LL + l;
    float ex = __expf(e[o]);
    att[o] = ex / (t == 0 ? 1.0f : acc);
    acc += ex;
  }
}

__global__ void rownorm_kernel(float* __restrict__ att, int rows, int Lr) {
  int wave = (int)((blockIdx.x * blockDim.x + threadIdx.x) >> 5);
  int lane = threadIdx.x & 31;
  if (wave >= rows) return;
  float* x = att + (size_t)wave * Lr;
  float s = 0.0f;
  for (int j = lane; j < Lr; j += 32) s += x[j];
  for (int o = 16; o; o >>= 1) s += __shfl_xor(s, o, 32);
  float inv = 1.0f / s;
  for (int j = lane; j < Lr; j += 32) x[j] *= inv;
}

// Masked softmax over previous steps; row 0 all-zero so dec_ctx GEMM yields 0.
__global__ void dec_softmax_kernel(const float* __restrict__ s, float* __restrict__ d, int rows) {
  int wave = (int)((blockIdx.x * blockDim.x + threadIdx.x) >> 5);
  int lane = threadIdx.x & 31;
  if (wave >= rows) return;
  int t = wave % TT;
  const float* x = s + (size_t)wave * TT;
  float* y = d + (size_t)wave * TT;
  if (t == 0) { for (int j = lane; j < TT; j += 32) y[j] = 0.0f; return; }
  float m = -1e30f;
  for (int j = lane; j < TT; j += 32) if (j < t) m = fmaxf(m, x[j]);
  for (int o = 16; o; o >>= 1) m = fmaxf(m, __shfl_xor(m, o, 32));
  float sum = 0.0f;
  for (int j = lane; j < TT; j += 32) if (j < t) sum += __expf(x[j] - m);
  for (int o = 16; o; o >>= 1) sum += __shfl_xor(sum, o, 32);
  float inv = 1.0f / sum;
  for (int j = lane; j < TT; j += 32) y[j] = (j < t) ? __expf(x[j] - m) * inv : 0.0f;
}

__global__ void copy_h_kernel(const float* __restrict__ dec, float* __restrict__ cat) {
  int idx = blockIdx.x * blockDim.x + threadIdx.x;
  if (idx >= BT * H2) return;
  int r = idx / H2, h = idx % H2;
  cat[(size_t)r * CATW + h] = dec[idx];
}

__global__ void ptr_gate_kernel(const float* __restrict__ cat,
                                const float* __restrict__ w,
                                const float* __restrict__ b,
                                float* __restrict__ p, int rows) {
  int wave = (int)((blockIdx.x * blockDim.x + threadIdx.x) >> 5);
  int lane = threadIdx.x & 31;
  if (wave >= rows) return;
  const float* r = cat + (size_t)wave * CATW;
  float acc = 0.0f;
  for (int k = lane; k < CATW; k += 32) acc += r[k] * w[k];
  for (int o = 16; o; o >>= 1) acc += __shfl_xor(acc, o, 32);
  if (lane == 0) p[wave] = 1.0f / (1.0f + __expf(-(acc + b[0])));
}

// Online softmax stats over logits living in d_out[:, :V] (row pitch VEXT)
__global__ void vocab_stats_kernel(const float* __restrict__ out,
                                   float* __restrict__ rowm, float* __restrict__ rowsum) {
  __shared__ float sm[256], ss[256];
  int row = blockIdx.x, tid = threadIdx.x;
  const float* x = out + (size_t)row * VEXT;
  float m = -1e30f, s = 0.0f;
  for (int v = tid; v < VV; v += 256) {
    float val = x[v];
    if (val > m) { s *= __expf(m - val); m = val; }
    s += __expf(val - m);
  }
  sm[tid] = m; ss[tid] = s;
  __syncthreads();
  for (int str = 128; str > 0; str >>= 1) {
    if (tid < str) {
      float m2 = sm[tid + str], s2 = ss[tid + str];
      float mm = fmaxf(sm[tid], m2);
      ss[tid] = ss[tid] * __expf(sm[tid] - mm) + s2 * __expf(m2 - mm);
      sm[tid] = mm;
    }
    __syncthreads();
  }
  if (tid == 0) { rowm[row] = sm[0]; rowsum[row] = ss[0]; }
}

__global__ void vocab_finalize_kernel(float* __restrict__ out,
                                      const float* __restrict__ rowm,
                                      const float* __restrict__ rowsum,
                                      const float* __restrict__ p) {
  int row = blockIdx.x, tid = threadIdx.x;
  float m = rowm[row];
  float inv = (1.0f - p[row]) / rowsum[row];
  float* x = out + (size_t)row * VEXT;
  for (int v = tid; v < VEXT; v += 256)
    x[v] = (v < VV) ? __expf(x[v] - m) * inv : 0.0f;
}

// Deterministic pointer scatter: thread per (b,t) row, serial over L (rows
// disjoint -> no atomics, bitwise identical across graph replays).
__global__ void scatter_ptr_kernel(float* __restrict__ out,
                                   const float* __restrict__ att,
                                   const float* __restrict__ p,
                                   const int* __restrict__ ext) {
  int row = blockIdx.x * blockDim.x + threadIdx.x;   // b*T + t
  if (row >= BT) return;
  int b = row / TT;
  float pv = p[row];
  float* o = out + (size_t)row * VEXT;
  const float* a = att + (size_t)row * LL;
  const int* ev = ext + (size_t)b * LL;
  for (int l = 0; l < LL; ++l) o[ev[l]] += pv * a[l];
}

// ---------------------------------------------------------------------------
// Host side
// ---------------------------------------------------------------------------
static inline int wave_blocks(long long waves) { return (int)((waves + 3) / 4); } // 128 thr

extern "C" void kernel_launch(void* const* d_in, const int* in_sizes, int n_in,
                              void* d_out, int out_size, void* d_ws, size_t ws_size,
                              hipStream_t stream) {
  const float* dec    = (const float*)d_in[0];
  const float* enc    = (const float*)d_in[1];
  const float* W_enc  = (const float*)d_in[2];
  const float* W_dec  = (const float*)d_in[3];
  const float* W_proj = (const float*)d_in[4];
  const float* W_voc  = (const float*)d_in[5];
  const float* w_ptr  = (const float*)d_in[6];
  const float* b_ptr  = (const float*)d_in[7];
  const int*   ext    = (const int*)d_in[8];
  float* out = (float*)d_out;

  // ---- workspace carve (256B aligned) ----
  char* ws = (char*)d_ws;
  auto carve = [&](size_t bytes) { void* p = (void*)ws; ws += (bytes + 255) & ~(size_t)255; return p; };
  // fp32 intermediates
  float* hWe   = (float*)carve((size_t)BT * H2 * 4);
  float* hWd   = (float*)carve((size_t)BT * H2 * 4);
  float* ebuf  = (float*)carve((size_t)BT * LL * 4);
  float* att   = (float*)carve((size_t)BT * LL * 4);
  float* sbuf  = (float*)carve((size_t)BT * TT * 4);
  float* datt  = (float*)carve((size_t)BT * TT * 4);
  float* cat   = (float*)carve((size_t)BT * CATW * 4);
  float* proj  = (float*)carve((size_t)BT * EE * 4);
  float* pgate = (float*)carve((size_t)BT * 4);
  float* rowm  = (float*)carve((size_t)BT * 4);
  float* rowsm = (float*)carve((size_t)BT * 4);
  // tile counts / k-steps
  const int mT_BT = BT / 16;            // 96
  const int mT_T  = TT / 16;            // 3
  const int nT_H  = H2 / 16;            // 32
  const int nT_L  = LL / 16;            // 25
  const int nT_T  = TT / 16;            // 3
  const int nT_E  = EE / 16;            // 8
  const int nT_V  = VV / 16;            // 3125
  const int S_H   = H2 / 32;            // 16
  const int S_L   = (LL + 31) / 32;     // 13 (zero-padded 400->416)
  const int S_T   = (TT + 31) / 32;     // 2  (zero-padded 48->64)
  const int S_CAT = CATW / 32;          // 48
  const int S_E   = EE / 32;            // 4
  // packed bf16 fragments
  __bf16* WencP = (__bf16*)carve((size_t)nT_H * S_H * 512 * 2);
  __bf16* WdecP = (__bf16*)carve((size_t)nT_H * S_H * 512 * 2);
  __bf16* WprjP = (__bf16*)carve((size_t)nT_E * S_CAT * 512 * 2);
  __bf16* WvocP = (__bf16*)carve((size_t)nT_V * S_E * 512 * 2);
  __bf16* decA  = (__bf16*)carve((size_t)mT_BT * S_H * 512 * 2);
  __bf16* hWeA  = (__bf16*)carve((size_t)BB * mT_T * S_H * 512 * 2);
  __bf16* hWdA  = (__bf16*)carve((size_t)BB * mT_T * S_H * 512 * 2);
  __bf16* decTB = (__bf16*)carve((size_t)BB * nT_T * S_H * 512 * 2);
  __bf16* decB  = (__bf16*)carve((size_t)BB * nT_H * S_T * 512 * 2);
  __bf16* attA  = (__bf16*)carve((size_t)BB * mT_T * S_L * 512 * 2);
  __bf16* dattA = (__bf16*)carve((size_t)BB * mT_T * S_T * 512 * 2);
  __bf16* catA  = (__bf16*)carve((size_t)mT_BT * S_CAT * 512 * 2);
  __bf16* projA = (__bf16*)carve((size_t)mT_BT * S_E * 512 * 2);
  // big region reused: encTB (e-GEMM, 25*16 blocks/b) then encB (ctx, 32*13 blocks/b)
  size_t encBlocksMax = (size_t)((nT_L * S_H > nT_H * S_L) ? nT_L * S_H : nT_H * S_L);
  __bf16* encPack = (__bf16*)carve((size_t)BB * encBlocksMax * 512 * 2);

  auto packs = [&](long long total) { return (int)((total + 255) / 256); };

  // ---- 0) weight packs (one-touch; W_vocab bf16 = 12.8MB -> L2 resident) ----
  { long long t = (long long)nT_H * S_H * 512;
    pack_b_kernel<<<packs(t), 256, 0, stream>>>(W_enc, WencP, H2, H2, H2, 0, nT_H, S_H, t);
    pack_b_kernel<<<packs(t), 256, 0, stream>>>(W_dec, WdecP, H2, H2, H2, 0, nT_H, S_H, t); }
  { long long t = (long long)nT_E * S_CAT * 512;
    pack_b_kernel<<<packs(t), 256, 0, stream>>>(W_proj, WprjP, CATW, EE, EE, 0, nT_E, S_CAT, t); }
  { long long t = (long long)nT_V * S_E * 512;
    pack_b_kernel<<<packs(t), 256, 0, stream>>>(W_voc, WvocP, EE, VV, VV, 0, nT_V, S_E, t); }

  // ---- 1) hWe = dec@W_enc, hWd = dec@W_dec ----
  { long long t = (long long)mT_BT * S_H * 512;
    pack_a_kernel<<<packs(t), 256, 0, stream>>>(dec, decA, BT, H2, H2, 0, mT_BT, S_H, t); }
  gemm_wmma_kernel<<<wave_blocks((long long)mT_BT * nT_H), 128, 0, stream>>>(
      decA, WencP, hWe, mT_BT, nT_H, S_H, H2, 0, 1);
  gemm_wmma_kernel<<<wave_blocks((long long)mT_BT * nT_H), 128, 0, stream>>>(
      decA, WdecP, hWd, mT_BT, nT_H, S_H, H2, 0, 1);

  // ---- 2) e[b] = hWe[b] @ enc[b]^T ----
  { long long t = (long long)BB * mT_T * S_H * 512;
    pack_a_kernel<<<packs(t), 256, 0, stream>>>(hWe, hWeA, TT, H2, H2, (long long)TT * H2, mT_T, S_H, t); }
  { long long t = (long long)BB * nT_L * S_H * 512;
    pack_bT_kernel<<<packs(t), 256, 0, stream>>>(enc, encPack, H2, LL, H2, (long long)LL * H2, nT_L, S_H, t); }
  gemm_wmma_kernel<<<wave_blocks((long long)BB * mT_T * nT_L), 128, 0, stream>>>(
      hWeA, encPack, ebuf, mT_T, nT_L, S_H, LL, (long long)TT * LL, BB);

  // ---- 3) intra-temporal attention (fp32, sequential in t) ----
  temporal_kernel<<<(BB * LL + 255) / 256, 256, 0, stream>>>(ebuf, att);
  rownorm_kernel<<<wave_blocks(BT), 128, 0, stream>>>(att, BT, LL);

  // ---- 4) s[b] = hWd[b] @ dec[b]^T ; masked softmax ----
  { long long t = (long long)BB * mT_T * S_H * 512;
    pack_a_kernel<<<packs(t), 256, 0, stream>>>(hWd, hWdA, TT, H2, H2, (long long)TT * H2, mT_T, S_H, t); }
  { long long t = (long long)BB * nT_T * S_H * 512;
    pack_bT_kernel<<<packs(t), 256, 0, stream>>>(dec, decTB, H2, TT, H2, (long long)TT * H2, nT_T, S_H, t); }
  gemm_wmma_kernel<<<wave_blocks((long long)BB * mT_T * nT_T), 128, 0, stream>>>(
      hWdA, decTB, sbuf, mT_T, nT_T, S_H, TT, (long long)TT * TT, BB);
  dec_softmax_kernel<<<wave_blocks(BT), 128, 0, stream>>>(sbuf, datt, BT);

  // ---- 5) cat = [h | enc_ctx | dec_ctx] ----
  copy_h_kernel<<<(BT * H2 + 255) / 256, 256, 0, stream>>>(dec, cat);
  { long long t = (long long)BB * mT_T * S_L * 512;   // att as A (K=400, padded)
    pack_a_kernel<<<packs(t), 256, 0, stream>>>(att, attA, TT, LL, LL, (long long)TT * LL, mT_T, S_L, t); }
  { long long t = (long long)BB * nT_H * S_L * 512;   // enc row-major as B (reuses encPack)
    pack_b_kernel<<<packs(t), 256, 0, stream>>>(enc, encPack, LL, H2, H2, (long long)LL * H2, nT_H, S_L, t); }
  gemm_wmma_kernel<<<wave_blocks((long long)BB * mT_T * nT_H), 128, 0, stream>>>(
      attA, encPack, cat + H2, mT_T, nT_H, S_L, CATW, (long long)TT * CATW, BB);
  { long long t = (long long)BB * mT_T * S_T * 512;   // datt as A (K=48, padded)
    pack_a_kernel<<<packs(t), 256, 0, stream>>>(datt, dattA, TT, TT, TT, (long long)TT * TT, mT_T, S_T, t); }
  { long long t = (long long)BB * nT_H * S_T * 512;   // dec row-major as B
    pack_b_kernel<<<packs(t), 256, 0, stream>>>(dec, decB, TT, H2, H2, (long long)TT * H2, nT_H, S_T, t); }
  gemm_wmma_kernel<<<wave_blocks((long long)BB * mT_T * nT_H), 128, 0, stream>>>(
      dattA, decB, cat + 2 * H2, mT_T, nT_H, S_T, CATW, (long long)TT * CATW, BB);

  // ---- 6) copy gate + projection ----
  ptr_gate_kernel<<<wave_blocks(BT), 128, 0, stream>>>(cat, w_ptr, b_ptr, pgate, BT);
  { long long t = (long long)mT_BT * S_CAT * 512;
    pack_a_kernel<<<packs(t), 256, 0, stream>>>(cat, catA, BT, CATW, CATW, 0, mT_BT, S_CAT, t); }
  gemm_wmma_kernel<<<wave_blocks((long long)mT_BT * nT_E), 128, 0, stream>>>(
      catA, WprjP, proj, mT_BT, nT_E, S_CAT, EE, 0, 1);

  // ---- 7) vocab GEMM: logits straight into d_out[:, :V] (row pitch VEXT) ----
  { long long t = (long long)mT_BT * S_E * 512;
    pack_a_kernel<<<packs(t), 256, 0, stream>>>(proj, projA, BT, EE, EE, 0, mT_BT, S_E, t); }
  gemm_wmma_kernel<<<wave_blocks((long long)mT_BT * nT_V), 128, 0, stream>>>(
      projA, WvocP, out, mT_BT, nT_V, S_E, VEXT, 0, 1);

  // ---- 8) softmax in place, (1-p) scale, zero OOV tail, pointer scatter ----
  vocab_stats_kernel<<<BT, 256, 0, stream>>>(out, rowm, rowsm);
  vocab_finalize_kernel<<<BT, 256, 0, stream>>>(out, rowm, rowsm, pgate);
  scatter_ptr_kernel<<<(BT + 255) / 256, 256, 0, stream>>>(out, att, pgate, ext);
}
